// GIoULoss_27470610825358
// MI455X (gfx1250) — compile-verified, matrix operand order checked by reading, not run on hardware
//
#include <hip/hip_runtime.h>

// ---------------------------------------------------------------------------
// GIoU loss for 9-gon pred vs rotated-rect target, N items -> scalar mean.
// Kernel 1: one thread per item, faithful port of the JAX reference.
// Kernel 2: one wave32; final reduction through V_WMMA_F32_16X16X4_F32
//           (A = 64 fp32 partials, B = ones => D rows are row-sums).
// ---------------------------------------------------------------------------

#define CEPS 1e-9f

struct P2 { float x, y; };

__device__ __forceinline__ float cross3(const P2& a, const P2& b, const P2& c) {
    return (b.x - a.x) * (c.y - a.y) - (b.y - a.y) * (c.x - a.x);
}

template <int NV>
__device__ float poly_signed_area(const P2* p) {
    float s = 0.f;
    for (int i = 0; i < NV; ++i) {
        int j = (i + 1 == NV) ? 0 : (i + 1);
        s += p[i].x * p[j].y - p[j].x * p[i].y;
    }
    return 0.5f * s;
}

// Mirrors reference convex_hull_pad: O(K^3) edge test with -1e-7 tolerance,
// centroid-angle sort of hull points, pad tail with last hull point.
template <int K>
__device__ void convex_hull_pad(const P2* pts, P2* hull) {
    bool on_hull[K];
    for (int i = 0; i < K; ++i) {
        bool any_edge = false;
        for (int j = 0; j < K; ++j) {
            if (j == i) continue;
            bool all_left = true;
            for (int k = 0; k < K; ++k) {
                float cr = cross3(pts[i], pts[j], pts[k]);
                all_left = all_left && (cr >= -1e-7f);
            }
            any_edge = any_edge || all_left;
        }
        on_hull[i] = any_edge;
    }
    float sx = 0.f, sy = 0.f, sw = 0.f;
    for (int i = 0; i < K; ++i) {
        if (on_hull[i]) { sx += pts[i].x; sy += pts[i].y; sw += 1.f; }
    }
    float dn = fmaxf(sw, 1.f);
    float cx = sx / dn, cy = sy / dn;

    float key[K];
    int idx[K];
    for (int i = 0; i < K; ++i) {
        key[i] = on_hull[i] ? atan2f(pts[i].y - cy, pts[i].x - cx)
                            : __builtin_inff();
        idx[i] = i;
    }
    // selection sort ascending (inf keys sink to the tail)
    for (int i = 0; i < K - 1; ++i) {
        int best = i;
        for (int j = i + 1; j < K; ++j)
            if (key[idx[j]] < key[idx[best]]) best = j;
        int t = idx[i]; idx[i] = idx[best]; idx[best] = t;
    }
    int cnt = 0;
    for (int i = 0; i < K; ++i) cnt += on_hull[i] ? 1 : 0;
    int lastIdx = idx[(cnt > 0) ? (cnt - 1) : 0];
    P2 last = pts[lastIdx];
    for (int i = 0; i < K; ++i)
        hull[i] = (i < cnt) ? pts[idx[i]] : last;
}

// Mirrors reference clip_convex for a fixed 16-vertex subject and 4 clip edges:
// candidates [inter_i, q_i] per edge, keep flagged in order, pad with last kept.
__device__ void clip_convex16(P2* poly, const P2* clip4) {
    for (int e = 0; e < 4; ++e) {
        P2 a = clip4[e];
        P2 b = clip4[(e + 1) & 3];
        float ex = b.x - a.x, ey = b.y - a.y;
        float dp[16];
        for (int i = 0; i < 16; ++i)
            dp[i] = ex * (poly[i].y - a.y) - ey * (poly[i].x - a.x);
        P2 kept[16];
        P2 firstCand; firstCand.x = 0.f; firstCand.y = 0.f;
        int cnt = 0;
        for (int i = 0; i < 16; ++i) {
            int j = (i + 1) & 15;
            float dpi = dp[i], dqi = dp[j];
            bool in_p = dpi >= 0.f;
            bool in_q = dqi >= 0.f;
            float den = dpi - dqi;
            float tden = (fabsf(den) < CEPS) ? CEPS : den;
            float t = dpi / tden;
            P2 inter;
            inter.x = poly[i].x + t * (poly[j].x - poly[i].x);
            inter.y = poly[i].y + t * (poly[j].y - poly[i].y);
            if (i == 0) firstCand = inter;
            if (in_p != in_q) { if (cnt < 16) kept[cnt] = inter; cnt++; }
            if (in_q)         { if (cnt < 16) kept[cnt] = poly[j]; cnt++; }
        }
        int kc = (cnt > 16) ? 16 : cnt;
        P2 padv = (cnt == 0) ? firstCand : kept[kc - 1];
        for (int i = 0; i < 16; ++i)
            poly[i] = (i < cnt) ? kept[i] : padv;
    }
}

__global__ __launch_bounds__(256)
void giou_partial_kernel(const float* __restrict__ pred,
                         const float* __restrict__ target,
                         const float* __restrict__ weight,
                         float* __restrict__ partial, int n) {
    const int tid = threadIdx.x;
    const int gid = blockIdx.x * 256 + tid;
    float term = 0.f;
    if (gid < n) {
        const float* pr = pred + (size_t)gid * 18;
        const float* tg = target + (size_t)gid * 8;
        P2 P[9], T[4];
        for (int k = 0; k < 9; ++k) { P[k].x = pr[2 * k]; P[k].y = pr[2 * k + 1]; }
        for (int k = 0; k < 4; ++k) { T[k].x = tg[2 * k]; T[k].y = tg[2 * k + 1]; }

        float sa = poly_signed_area<4>(T);
        P2 Tc[4];
        if (sa < 0.f) { Tc[0] = T[3]; Tc[1] = T[2]; Tc[2] = T[1]; Tc[3] = T[0]; }
        else          { Tc[0] = T[0]; Tc[1] = T[1]; Tc[2] = T[2]; Tc[3] = T[3]; }

        P2 hullA[9];
        convex_hull_pad<9>(P, hullA);
        float areaA = fabsf(poly_signed_area<9>(hullA));
        float areaB = fabsf(sa);

        P2 poly[16];
        for (int i = 0; i < 9; ++i) poly[i] = hullA[i];
        for (int i = 9; i < 16; ++i) poly[i] = hullA[8];
        clip_convex16(poly, Tc);
        float inter = fabsf(poly_signed_area<16>(poly));
        float uni = areaA + areaB - inter;

        P2 PT[13];
        for (int i = 0; i < 9; ++i) PT[i] = P[i];
        for (int i = 0; i < 4; ++i) PT[9 + i] = T[i];
        P2 hullC[13];
        convex_hull_pad<13>(PT, hullC);
        float areaC = fabsf(poly_signed_area<13>(hullC));

        float giou = inter / fmaxf(uni, CEPS) - (areaC - uni) / fmaxf(areaC, CEPS);
        term = (1.f - giou) * weight[gid];
    }
    __shared__ float red[256];
    red[tid] = term;
    __syncthreads();
    for (int s = 128; s > 0; s >>= 1) {
        if (tid < s) red[tid] += red[tid + s];
        __syncthreads();
    }
    if (tid == 0) partial[blockIdx.x] = red[0];
}

typedef __attribute__((ext_vector_type(2)))  float    v2f;
typedef __attribute__((ext_vector_type(8)))  float    v8f;
typedef __attribute__((ext_vector_type(16))) _Float16 v16h;

// One wave32; EXEC is all-ones at the WMMA (required). Lanes pack fp32
// partial sums into the 16x4 fp32 A layout; B = ones so D rows = row-sums.
__global__ __launch_bounds__(32)
void giou_reduce_kernel(const float* __restrict__ partial,
                        float* __restrict__ out, int nPartial, float invN) {
    const int lane = threadIdx.x;  // 0..31, wave32
    float a0 = 0.f, a1 = 0.f;
    for (int i = lane; i < nPartial; i += 64) a0 += partial[i];
    for (int i = lane + 32; i < nPartial; i += 64) a1 += partial[i];

    v8f C = {};
    v8f D;
#if __has_builtin(__builtin_amdgcn_wmma_f32_16x16x4_f32)
    v2f A; A.x = a0; A.y = a1;
    v2f B; B.x = 1.f; B.y = 1.f;
    D = __builtin_amdgcn_wmma_f32_16x16x4_f32(
        false, A, false, B, (short)0, C, false, false);
#else
    // fallback: f16 WMMA (codegen-confirmed) with partials in A[0..1]
    v16h A = {};
    A[0] = (_Float16)a0; A[1] = (_Float16)a1;
    v16h B = {(_Float16)1.f, (_Float16)1.f, (_Float16)1.f, (_Float16)1.f,
              (_Float16)1.f, (_Float16)1.f, (_Float16)1.f, (_Float16)1.f,
              (_Float16)1.f, (_Float16)1.f, (_Float16)1.f, (_Float16)1.f,
              (_Float16)1.f, (_Float16)1.f, (_Float16)1.f, (_Float16)1.f};
    D = __builtin_amdgcn_wmma_f32_16x16x32_f16(
        false, A, false, B, (short)0, C, false, false);
#endif
    // lanes 0..15 hold D rows M=0..7 (all columns equal), lanes 16..31 rows 8..15
    float s = D[0] + D[1] + D[2] + D[3] + D[4] + D[5] + D[6] + D[7];
    float tot = s + __shfl_xor(s, 16, 32);
    if (lane == 0) out[0] = tot * invN;
}

extern "C" void kernel_launch(void* const* d_in, const int* in_sizes, int n_in,
                              void* d_out, int out_size, void* d_ws, size_t ws_size,
                              hipStream_t stream) {
    (void)n_in; (void)out_size; (void)ws_size;
    const float* pred   = (const float*)d_in[0];
    const float* target = (const float*)d_in[1];
    const float* weight = (const float*)d_in[2];
    float* out = (float*)d_out;
    float* partial = (float*)d_ws;

    const int n = in_sizes[2];            // N items (weight count)
    const int blocks = (n + 255) / 256;   // 128 for N=32768

    giou_partial_kernel<<<blocks, 256, 0, stream>>>(pred, target, weight, partial, n);
    giou_reduce_kernel<<<1, 32, 0, stream>>>(partial, out, blocks, 1.0f / (float)n);
}
